// DecoderLayer_60060822667509
// MI455X (gfx1250) — compile-verified, matrix operand order checked by reading, not run on hardware
//
#include <hip/hip_runtime.h>

typedef unsigned long long ull;
typedef __attribute__((ext_vector_type(16))) __bf16 v16bf;
typedef __attribute__((ext_vector_type(8)))  __bf16 bf16x8;
typedef __attribute__((ext_vector_type(8)))  float v8f;

#define AS1 __attribute__((address_space(1)))
typedef const AS1 __bf16* gbf;
typedef AS1 __bf16*       gbfw;
typedef const AS1 float*  gfl;
typedef AS1 float*        gflw;

#define DM  512
#define RW  4096   // B*SLEN
#define SEQ 1024

// ---------------- descriptors (device-built each launch) ----------------
struct GemmDesc {
  ull A, W, bias, addsrc, res, outF, outB;   // absolute device addresses (0 == null)
  float scale, addscale, resscale;
  int act;       // 0 none, 1 relu, 2 gelu
  int enabled;
};
struct EwDesc {
  ull x0, x1, x2, g, b, outF, outB;
  float scale;
  int op;        // 0 LN(sum), 1 sum, 2 x0*sigmoid(x1)+x2, 3 x0*x1
  int enabled;
};
struct PlanArgs {
  const float *inpute, *inputo, *node_p, *edge_p;
  const float *edge_b, *edge_g, *edge_beta;
  const float *node_b, *node_g, *node_beta;
  char* ws;
};

// ---------------- workspace layout ----------------
constexpr size_t SZF = (size_t)RW * DM * 4;   // fp32 activation tensor
constexpr size_t SZB = (size_t)RW * DM * 2;   // bf16 activation tensor
constexpr size_t OFF_EDGEWT = 0;                                   // 34 x [N][K] bf16
constexpr size_t OFF_NODEWT = OFF_EDGEWT + 34ull * DM * DM * 2;    // 32 x [N][K] bf16
constexpr size_t OFF_ARENA  = OFF_NODEWT + 32ull * DM * DM * 2;    // outs[0..7] fp32
constexpr size_t OFF_PATHF  = OFF_ARENA + 8 * SZF;                 // q/k/v fp32
constexpr size_t OFF_PATHB  = OFF_PATHF + 3 * SZF;                 // q/k/v bf16
constexpr size_t OFF_STAGE  = OFF_PATHB + 3 * SZB;                 // LN-staged A bf16 x3
constexpr size_t OFF_QN     = OFF_STAGE + 3 * SZB;                 // LN(q) bf16
constexpr size_t OFF_T0     = OFF_QN + SZB;                        // temp fp32
constexpr size_t OFF_T1     = OFF_T0 + SZF;                        // temp fp32
constexpr size_t OFF_T2B    = OFF_T1 + SZF;                        // temp bf16
constexpr size_t OFF_QH     = OFF_T2B + SZB;
constexpr size_t OFF_KH     = OFF_QH + SZB;
constexpr size_t OFF_VH     = OFF_KH + SZB;
constexpr size_t OFF_VT     = OFF_VH + SZB;                        // V transposed [B][512][1024]
constexpr size_t OFF_OB     = OFF_VT + SZB;                        // attention out bf16
constexpr size_t OFF_GD     = OFF_OB + SZB;                        // GemmDesc[8][7]
constexpr size_t OFF_ED     = OFF_GD + 56 * sizeof(GemmDesc);      // EwDesc[8][5]
constexpr size_t OFF_ACT    = OFF_ED + 40 * sizeof(EwDesc);        // int[8]
constexpr size_t OFF_FMASK  = OFF_ACT + 8 * sizeof(int);           // unsigned

// ---------------- helpers ----------------
__device__ __forceinline__ float geluf(float x) {
  return 0.5f * x * (1.f + tanhf(0.7978845608f * (x + 0.044715f * x * x * x)));
}

// 16-bit fragment loader, CDNA5 wave32 A/B layout (16x32 / 32x16), row-major global src.
__device__ __forceinline__ v16bf load_frag(gbf base, int ld, int row0, int k0, int lane) {
  int m   = lane & 15;
  int klo = (lane & 16) ? 8 : 0;
  gbf p = base + (size_t)(row0 + m) * ld + k0 + klo;
  bf16x8 lo = *(const AS1 bf16x8*)p;
  bf16x8 hi = *(const AS1 bf16x8*)(p + 16);
  v16bf f;
#pragma unroll
  for (int i = 0; i < 8; i++) { f[i] = lo[i]; f[i + 8] = hi[i]; }
  return f;
}

#define WMMA_BF16(a, b, c) __builtin_amdgcn_wmma_f32_16x16x32_bf16(false, (a), false, (b), (short)0, (c), false, false)

__device__ static int amaxm(const float* v, int n, unsigned mask) {
  int bi = 0; float bv = -3.4e38f;
  for (int j = 0; j < n; j++) {
    if ((mask >> j) & 1) continue;
    if (v[j] > bv) { bv = v[j]; bi = j; }
  }
  return bi;
}
__device__ static float selwf(const float* v, int n, int sel, unsigned mask) {
  float mx = -3.4e38f;
  for (int j = 0; j < n; j++) if (!((mask >> j) & 1) && v[j] > mx) mx = v[j];
  float s = 0.f;
  for (int j = 0; j < n; j++) if (!((mask >> j) & 1)) s += expf(v[j] - mx);
  return expf(v[sel] - mx) / s;
}

// ---------------- weight transpose+convert: dst[n][k] = (bf16)src[k][n] ----------------
__global__ __launch_bounds__(256) void wconv_kernel(const float* src, __bf16* dst) {
  __shared__ float tile[32][33];
  int mat = blockIdx.z;
  gfl  S = (gfl)(ull)(src + (size_t)mat * DM * DM);
  gbfw D = (gbfw)(ull)(dst + (size_t)mat * DM * DM);
  int n0 = blockIdx.x * 32, k0 = blockIdx.y * 32;
  for (int i = threadIdx.y; i < 32; i += 8)
    tile[i][threadIdx.x] = S[(size_t)(k0 + i) * DM + n0 + threadIdx.x];
  __syncthreads();
  for (int i = threadIdx.y; i < 32; i += 8)
    D[(size_t)(n0 + i) * DM + k0 + threadIdx.x] = (__bf16)tile[threadIdx.x][i];
}

// ---------------- device-side routing + descriptor planning ----------------
__global__ void plan_kernel(PlanArgs pa) {
  if (threadIdx.x || blockIdx.x) return;
  char* ws = pa.ws;
  GemmDesc* GD = (GemmDesc*)(ws + OFF_GD);
  EwDesc*  ED = (EwDesc*)(ws + OFF_ED);
  int* ACT = (int*)(ws + OFF_ACT);
  for (int i = 0; i < 56; i++) GD[i].enabled = 0;
  for (int i = 0; i < 40; i++) ED[i].enabled = 0;

  unsigned processed = 0;
  int lind = 0;
  for (int c = 0; c < 8; c++) {
    int nsrc = (c + 2 < 5) ? c + 2 : 5;
    int snode = c - nsrc;
    int L = nsrc * 5;
    const float* np = pa.node_p + c * 8;
    const float* eq = pa.edge_p + 0 * 170 + lind * 5;
    const float* ek = pa.edge_p + 1 * 170 + lind * 5;
    const float* ev = pa.edge_p + 2 * 170 + lind * 5;

    int act = amaxm(np, 8, 0u);
    float aw = selwf(np, 8, act, 0u);
    unsigned qm = 0x1Fu;                        // q path: source 0 masked
    int qsel = amaxm(eq, L, qm);
    float qw = selwf(eq, L, qsel, qm);
    int ksel = -1, vsel = -1; float kw = 0.f, vw = 0.f;
    if (act < 7) {
      unsigned km = (act > 0) ? qm : 0u;
      ksel = amaxm(ek, L, km);
      kw = selwf(ek, L, ksel, km);
      int ktype = (ksel / 5 == 0) ? -2 : -1;
      if (act < 5) {
        if (act == 0 && ktype == -2) { vsel = amaxm(ev, 5, 0u); vw = selwf(ev, 5, vsel, 0u); }
        else                         { vsel = amaxm(ev, L, km); vw = selwf(ev, L, vsel, km); }
      }
    }

    auto P = [&](size_t off) { return (ull)(ws + off); };
    auto setew = [&](int slot, ull x0, ull x1, ull x2, ull g, ull b, float sc, ull oF, ull oB, int op) {
      EwDesc& e = ED[c * 5 + slot];
      e.x0 = x0; e.x1 = x1; e.x2 = x2; e.g = g; e.b = b; e.scale = sc;
      e.outF = oF; e.outB = oB; e.op = op; e.enabled = 1;
    };
    auto setgm = [&](int slot, ull A, ull W, ull bias, ull add, float addsc, int gact,
                     float sc, ull res, float ressc, ull oF, ull oB) {
      GemmDesc& g = GD[c * 7 + slot];
      g.A = A; g.W = W; g.bias = bias; g.addsrc = add; g.addscale = addsc; g.act = gact;
      g.scale = sc; g.res = res; g.resscale = ressc; g.outF = oF; g.outB = oB; g.enabled = 1;
    };
    auto dopath = [&](int p, int sel, float wsel) {
      if (sel < 0) return;
      int se = sel / 5, op = sel % 5, e = lind + se;
      int inn = (se == 0) ? -2 : snode + se;
      if (inn >= 0) processed |= (1u << inn);
      ull x = (inn == -2) ? (ull)pa.inpute : (inn == -1) ? (ull)pa.inputo
                          : P(OFF_ARENA + (size_t)inn * SZF);
      ull pF = P(OFF_PATHF + (size_t)p * SZF);
      ull pB = P(OFF_PATHB + (size_t)p * SZB);
      ull st = P(OFF_STAGE + (size_t)p * SZB);
      if (op == 4) {                                   // identity: scaled copy
        setew(p, x, 0, 0, 0, 0, wsel, pF, pB, 1);
      } else {
        if (op == 3) setew(p, x, 0, 0, 0, 0, 1.f, 0, st, 1);                 // plain linear: no LN
        else setew(p, x, 0, 0, (ull)(pa.edge_g + e * DM), (ull)(pa.edge_beta + e * DM),
                   1.f, 0, st, 0);                                            // LN stage
        int ga = (op == 0) ? 1 : (op == 1) ? 2 : 0;
        setgm(p, st, P(OFF_EDGEWT + (size_t)e * DM * DM * 2), (ull)(pa.edge_b + e * DM),
              0, 0.f, ga, wsel, 0, 0.f, pF, pB);
      }
    };
    dopath(0, qsel, qw);
    dopath(1, ksel, kw);
    dopath(2, vsel, vw);

    ull qF = P(OFF_PATHF), kF = P(OFF_PATHF + SZF), vF = P(OFF_PATHF + 2 * SZF);
    ull qB = P(OFF_PATHB), kB = P(OFF_PATHB + SZB), vB = P(OFF_PATHB + 2 * SZB);
    ull outc = P(OFF_ARENA + (size_t)c * SZF);
    ull ng = (ull)(pa.node_g + c * DM), nbe = (ull)(pa.node_beta + c * DM);
    ull W0 = P(OFF_NODEWT + ((size_t)c * 4 + 0) * DM * DM * 2);
    ull W1 = P(OFF_NODEWT + ((size_t)c * 4 + 1) * DM * DM * 2);
    ull W2 = P(OFF_NODEWT + ((size_t)c * 4 + 2) * DM * DM * 2);
    ull W3 = P(OFF_NODEWT + ((size_t)c * 4 + 3) * DM * DM * 2);
    ull b0 = (ull)(pa.node_b + ((size_t)c * 4 + 0) * DM);
    ull b1 = (ull)(pa.node_b + ((size_t)c * 4 + 1) * DM);
    ull b2 = (ull)(pa.node_b + ((size_t)c * 4 + 2) * DM);
    ull b3 = (ull)(pa.node_b + ((size_t)c * 4 + 3) * DM);

    switch (act) {
      case 0:  // MHA
        setew(3, qF, 0, 0, ng, nbe, 1.f, 0, P(OFF_QN), 0);
        setgm(3, P(OFF_QN), W0, b0, 0, 0.f, 0, 1.f, 0, 0.f, 0, P(OFF_QH));
        setgm(4, kB, W1, b1, 0, 0.f, 0, 1.f, 0, 0.f, 0, P(OFF_KH));
        setgm(5, vB, W2, b2, 0, 0.f, 0, 1.f, 0, 0.f, 0, P(OFF_VH));
        setgm(6, P(OFF_OB), W3, b3, 0, 0.f, 0, aw, qF, aw, outc, 0);
        break;
      case 1:  // q + (gelu(qW0+b0)*(kW1+b1))W3 + b3
        setgm(3, qB, W0, b0, 0, 0.f, 2, 1.f, 0, 0.f, P(OFF_T0), 0);
        setgm(4, kB, W1, b1, 0, 0.f, 0, 1.f, 0, 0.f, P(OFF_T1), 0);
        setew(4, P(OFF_T0), P(OFF_T1), 0, 0, 0, 1.f, 0, P(OFF_T2B), 3);
        setgm(6, P(OFF_T2B), W3, b3, 0, 0.f, 0, aw, qF, aw, outc, 0);
        break;
      case 2: setew(3, qF, kF, vF, ng, nbe, aw, outc, 0, 0); break;  // LN(q+k+v)
      case 3:  // q + relu(qW0+kW1+vW2)W3 + b3
        setgm(3, qB, W0, 0, 0, 0.f, 0, 1.f, 0, 0.f, P(OFF_T0), 0);
        setgm(4, kB, W1, 0, P(OFF_T0), 1.f, 0, 1.f, 0, 0.f, P(OFF_T0), 0);
        setgm(5, vB, W2, 0, P(OFF_T0), 1.f, 1, 1.f, 0, 0.f, 0, P(OFF_T2B));
        setgm(6, P(OFF_T2B), W3, b3, 0, 0.f, 0, aw, qF, aw, outc, 0);
        break;
      case 4: setew(3, qF, kF, vF, 0, 0, aw, outc, 0, 2); break;     // q*sigmoid(k)+v
      case 5: setgm(4, kB, W1, b1, 0, 0.f, 2, aw, qF, aw, outc, 0); break; // q+gelu(kW1+b1)
      case 6: setew(3, qF, kF, 0, 0, 0, aw, outc, 0, 1); break;      // q+k
      default: setew(3, qF, 0, 0, ng, nbe, aw, outc, 0, 0); break;   // LN(q)
    }
    ACT[c] = act;
    lind += nsrc;
  }
  *(unsigned*)(ws + OFF_FMASK) = (~processed) & 0xFFu;
}

// ---------------- generic bf16 WMMA GEMM: [4096x512] @ [512x512] ----------------
__global__ __launch_bounds__(256) void gemm_bf16_kernel(char* ws, int c, int slot) {
  GemmDesc d = ((const GemmDesc*)(ws + OFF_GD))[c * 7 + slot];
  if (!d.enabled) return;
  gbf A = (gbf)d.A;
  gbf W = (gbf)d.W;     // [N][K] (pre-transposed)
  int lane = threadIdx.x & 31, wv = threadIdx.x >> 5;
  int row0 = (blockIdx.x * 8 + wv) * 16;
  int col0 = blockIdx.y * 64;
  v8f z = {0.f, 0.f, 0.f, 0.f, 0.f, 0.f, 0.f, 0.f};
  v8f acc0 = z, acc1 = z, acc2 = z, acc3 = z;
  for (int kk = 0; kk < DM; kk += 32) {
    v16bf a  = load_frag(A, DM, row0, kk, lane);
    v16bf w0 = load_frag(W, DM, col0,      kk, lane);
    v16bf w1 = load_frag(W, DM, col0 + 16, kk, lane);
    v16bf w2 = load_frag(W, DM, col0 + 32, kk, lane);
    v16bf w3 = load_frag(W, DM, col0 + 48, kk, lane);
    acc0 = WMMA_BF16(a, w0, acc0);
    acc1 = WMMA_BF16(a, w1, acc1);
    acc2 = WMMA_BF16(a, w2, acc2);
    acc3 = WMMA_BF16(a, w3, acc3);
  }
  int n = lane & 15, m0 = (lane & 16) ? 8 : 0;
  gfl bias = (gfl)d.bias;
  gfl add  = (gfl)d.addsrc;
  gfl res  = (gfl)d.res;
  gflw oF  = (gflw)d.outF;
  gbfw oB  = (gbfw)d.outB;
  v8f accs[4] = {acc0, acc1, acc2, acc3};
#pragma unroll
  for (int j = 0; j < 4; j++) {
    int col = col0 + j * 16 + n;
    float bv = bias ? bias[col] : 0.f;
#pragma unroll
    for (int r = 0; r < 8; r++) {
      size_t idx = (size_t)(row0 + m0 + r) * DM + col;
      float v = accs[j][r] + bv;
      if (add) v += d.addscale * add[idx];
      if (d.act == 1) v = fmaxf(v, 0.f);
      else if (d.act == 2) v = geluf(v);
      v *= d.scale;
      if (res) v += d.resscale * res[idx];
      if (oF) oF[idx] = v;
      if (oB) oB[idx] = (__bf16)v;
    }
  }
}

// ---------------- elementwise / LayerNorm kernel ----------------
__global__ __launch_bounds__(256) void ew_kernel(char* ws, int c, int slot) {
  __shared__ float sm[16];
  EwDesc d = ((const EwDesc*)(ws + OFF_ED))[c * 5 + slot];
  if (!d.enabled) return;
  size_t base = (size_t)blockIdx.x * DM;
  gfl x0 = (gfl)d.x0;
  gfl x1 = (gfl)d.x1;
  gfl x2 = (gfl)d.x2;
  float v[2];
#pragma unroll
  for (int i = 0; i < 2; i++) {
    size_t idx = base + threadIdx.x + i * 256;
    float a0 = x0[idx];
    float a1 = x1 ? x1[idx] : 0.f;
    float a2 = x2 ? x2[idx] : 0.f;
    float r;
    if (d.op == 2)      r = a0 / (1.f + __expf(-a1)) + a2;
    else if (d.op == 3) r = a0 * a1;
    else                r = a0 + a1 + a2;
    v[i] = r;
  }
  float y0, y1;
  if (d.op == 0) {
    float s = v[0] + v[1], ss = v[0] * v[0] + v[1] * v[1];
#pragma unroll
    for (int off = 16; off > 0; off >>= 1) { s += __shfl_xor(s, off, 32); ss += __shfl_xor(ss, off, 32); }
    int wv = threadIdx.x >> 5;
    if ((threadIdx.x & 31) == 0) { sm[wv] = s; sm[8 + wv] = ss; }
    __syncthreads();
    float ts = 0.f, tss = 0.f;
#pragma unroll
    for (int i = 0; i < 8; i++) { ts += sm[i]; tss += sm[8 + i]; }
    float mean = ts * (1.f / DM);
    float var  = tss * (1.f / DM) - mean * mean;
    float inv = rsqrtf(var + 1e-6f);
    gfl g = (gfl)d.g;
    gfl b = (gfl)d.b;
    int c0 = threadIdx.x, c1 = threadIdx.x + 256;
    y0 = ((v[0] - mean) * inv * g[c0] + b[c0]) * d.scale;
    y1 = ((v[1] - mean) * inv * g[c1] + b[c1]) * d.scale;
  } else {
    y0 = v[0] * d.scale; y1 = v[1] * d.scale;
  }
  gflw oF = (gflw)d.outF;
  gbfw oB = (gbfw)d.outB;
  size_t i0 = base + threadIdx.x, i1 = i0 + 256;
  if (oF) { oF[i0] = y0; oF[i1] = y1; }
  if (oB) { oB[i0] = (__bf16)y0; oB[i1] = (__bf16)y1; }
}

// ---------------- V transpose for attention (bf16) ----------------
__global__ __launch_bounds__(256) void vtrans_kernel(char* ws, int c) {
  if (((const int*)(ws + OFF_ACT))[c] != 0) return;
  gbf  vh = (gbf)(ull)(ws + OFF_VH);
  gbfw vt = (gbfw)(ull)(ws + OFF_VT);
  __shared__ __bf16 tile[32][33];
  int b = blockIdx.z;
  int s0 = blockIdx.x * 32, c0 = blockIdx.y * 32;
  for (int i = threadIdx.y; i < 32; i += 8)
    tile[i][threadIdx.x] = vh[(size_t)(b * SEQ + s0 + i) * DM + c0 + threadIdx.x];
  __syncthreads();
  for (int i = threadIdx.y; i < 32; i += 8)
    vt[((size_t)b * DM + c0 + i) * SEQ + s0 + threadIdx.x] = tile[threadIdx.x][i];
}

// ---------------- flash attention (8 heads, dh=64, S=1024) ----------------
__global__ __launch_bounds__(256) void attn_kernel(char* ws, int c) {
  if (((const int*)(ws + OFF_ACT))[c] != 0) return;
  gbf  qh = (gbf)(ull)(ws + OFF_QH);
  gbf  kh = (gbf)(ull)(ws + OFF_KH);
  gbf  vt = (gbf)(ull)(ws + OFF_VT);
  gbfw ob = (gbfw)(ull)(ws + OFF_OB);
  __shared__ __attribute__((aligned(16))) __bf16 psh[8][16][32];
  int lane = threadIdx.x & 31, wv = threadIdx.x >> 5;
  int job = blockIdx.x * 8 + wv;            // 2048 jobs = B(4) x H(8) x qtiles(64)
  int qt = job & 63, h = (job >> 6) & 7, b = job >> 9;
  int qrow0 = b * SEQ + qt * 16;
  int n = lane & 15, m0 = (lane & 16) ? 8 : 0, klo = (lane & 16) ? 8 : 0;
  v16bf qa0 = load_frag(qh, DM, qrow0, h * 64, lane);
  v16bf qa1 = load_frag(qh, DM, qrow0, h * 64 + 32, lane);
  v8f z = {0.f, 0.f, 0.f, 0.f, 0.f, 0.f, 0.f, 0.f};
  v8f o0 = z, o1 = z, o2 = z, o3 = z;
  float rm[8], rl[8];
#pragma unroll
  for (int r = 0; r < 8; r++) { rm[r] = -1e30f; rl[r] = 0.f; }

  for (int kt = 0; kt < SEQ; kt += 32) {
    v8f s0 = z, s1 = z;
    v16bf kb0 = load_frag(kh, DM, b * SEQ + kt,      h * 64,      lane);
    v16bf kb1 = load_frag(kh, DM, b * SEQ + kt,      h * 64 + 32, lane);
    s0 = WMMA_BF16(qa0, kb0, s0);  s0 = WMMA_BF16(qa1, kb1, s0);
    v16bf kc0 = load_frag(kh, DM, b * SEQ + kt + 16, h * 64,      lane);
    v16bf kc1 = load_frag(kh, DM, b * SEQ + kt + 16, h * 64 + 32, lane);
    s1 = WMMA_BF16(qa0, kc0, s1);  s1 = WMMA_BF16(qa1, kc1, s1);

    float corr[8];
#pragma unroll
    for (int r = 0; r < 8; r++) {
      float a0 = s0[r] * 0.125f, a1 = s1[r] * 0.125f;
      float mx = fmaxf(a0, a1);
#pragma unroll
      for (int off = 8; off > 0; off >>= 1) mx = fmaxf(mx, __shfl_xor(mx, off, 32));
      float nm = fmaxf(rm[r], mx);
      float p0 = __expf(a0 - nm), p1 = __expf(a1 - nm);
      float rs = p0 + p1;
#pragma unroll
      for (int off = 8; off > 0; off >>= 1) rs += __shfl_xor(rs, off, 32);
      float cr = __expf(rm[r] - nm);
      rl[r] = rl[r] * cr + rs;
      rm[r] = nm;
      corr[r] = cr;
      psh[wv][m0 + r][n]      = (__bf16)p0;
      psh[wv][m0 + r][n + 16] = (__bf16)p1;
    }
#pragma unroll
    for (int r = 0; r < 8; r++) { o0[r] *= corr[r]; o1[r] *= corr[r]; o2[r] *= corr[r]; o3[r] *= corr[r]; }

    asm volatile("s_wait_dscnt 0" ::: "memory");   // C-layout -> A-layout via per-wave LDS
    bf16x8 plo = *(const bf16x8*)&psh[wv][lane & 15][klo];
    bf16x8 phi = *(const bf16x8*)&psh[wv][lane & 15][16 + klo];
    v16bf pa;
#pragma unroll
    for (int i = 0; i < 8; i++) { pa[i] = plo[i]; pa[i + 8] = phi[i]; }

    v16bf vb0 = load_frag(vt, SEQ, b * DM + h * 64,      kt, lane);
    v16bf vb1 = load_frag(vt, SEQ, b * DM + h * 64 + 16, kt, lane);
    v16bf vb2 = load_frag(vt, SEQ, b * DM + h * 64 + 32, kt, lane);
    v16bf vb3 = load_frag(vt, SEQ, b * DM + h * 64 + 48, kt, lane);
    o0 = WMMA_BF16(pa, vb0, o0);
    o1 = WMMA_BF16(pa, vb1, o1);
    o2 = WMMA_BF16(pa, vb2, o2);
    o3 = WMMA_BF16(pa, vb3, o3);
  }
#pragma unroll
  for (int r = 0; r < 8; r++) {
    float inv = 1.f / rl[r];
    size_t off = (size_t)(qrow0 + m0 + r) * DM + h * 64 + n;
    ob[off]      = (__bf16)(o0[r] * inv);
    ob[off + 16] = (__bf16)(o1[r] * inv);
    ob[off + 32] = (__bf16)(o2[r] * inv);
    ob[off + 48] = (__bf16)(o3[r] * inv);
  }
}

// ---------------- final: masked sum of unconsumed outs + LN ----------------
__global__ __launch_bounds__(256) void final_kernel(char* ws, const float* g, const float* b, float* out) {
  __shared__ float sm[16];
  unsigned mask = *(const unsigned*)(ws + OFF_FMASK);
  size_t base = (size_t)blockIdx.x * DM;
  gfl arena = (gfl)(ull)(ws + OFF_ARENA);
  float v[2];
#pragma unroll
  for (int i = 0; i < 2; i++) {
    size_t idx = base + threadIdx.x + i * 256;
    float s = 0.f;
    for (int nn = 0; nn < 8; nn++)
      if ((mask >> nn) & 1) s += arena[(size_t)nn * RW * DM + idx];
    v[i] = s;
  }
  float s = v[0] + v[1], ss = v[0] * v[0] + v[1] * v[1];
#pragma unroll
  for (int off = 16; off > 0; off >>= 1) { s += __shfl_xor(s, off, 32); ss += __shfl_xor(ss, off, 32); }
  int wv = threadIdx.x >> 5;
  if ((threadIdx.x & 31) == 0) { sm[wv] = s; sm[8 + wv] = ss; }
  __syncthreads();
  float ts = 0.f, tss = 0.f;
#pragma unroll
  for (int i = 0; i < 8; i++) { ts += sm[i]; tss += sm[8 + i]; }
  float mean = ts * (1.f / DM);
  float var  = tss * (1.f / DM) - mean * mean;
  float inv = rsqrtf(var + 1e-6f);
  gfl gg = (gfl)(ull)g;
  gfl bb = (gfl)(ull)b;
  gflw oo = (gflw)(ull)out;
  int c0 = threadIdx.x, c1 = threadIdx.x + 256;
  oo[base + c0] = (v[0] - mean) * inv * gg[c0] + bb[c0];
  oo[base + c1] = (v[1] - mean) * inv * gg[c1] + bb[c1];
}

// ---------------- host launch ----------------
extern "C" void kernel_launch(void* const* d_in, const int* in_sizes, int n_in,
                              void* d_out, int out_size, void* d_ws, size_t ws_size,
                              hipStream_t stream) {
  const float* inpute    = (const float*)d_in[0];
  const float* inputo    = (const float*)d_in[1];
  const float* node_p    = (const float*)d_in[2];
  const float* edge_p    = (const float*)d_in[3];
  const float* edge_W    = (const float*)d_in[4];
  const float* edge_b    = (const float*)d_in[5];
  const float* edge_g    = (const float*)d_in[6];
  const float* edge_beta = (const float*)d_in[7];
  const float* node_W    = (const float*)d_in[8];
  const float* node_b    = (const float*)d_in[9];
  const float* node_g    = (const float*)d_in[10];
  const float* node_beta = (const float*)d_in[11];
  const float* out_g     = (const float*)d_in[12];
  const float* out_beta  = (const float*)d_in[13];
  char* ws = (char*)d_ws;

  dim3 tb(32, 8);
  wconv_kernel<<<dim3(16, 16, 34), tb, 0, stream>>>(edge_W, (__bf16*)(ws + OFF_EDGEWT));
  wconv_kernel<<<dim3(16, 16, 32), tb, 0, stream>>>(node_W, (__bf16*)(ws + OFF_NODEWT));

  PlanArgs pa;
  pa.inpute = inpute; pa.inputo = inputo; pa.node_p = node_p; pa.edge_p = edge_p;
  pa.edge_b = edge_b; pa.edge_g = edge_g; pa.edge_beta = edge_beta;
  pa.node_b = node_b; pa.node_g = node_g; pa.node_beta = node_beta;
  pa.ws = ws;
  plan_kernel<<<1, 1, 0, stream>>>(pa);

  dim3 gg(32, 8);   // 4096/(16*8) x 512/64
  for (int c = 0; c < 8; c++) {
    for (int p = 0; p < 3; p++) ew_kernel<<<RW, 256, 0, stream>>>(ws, c, p);        // edge pre (LN/copy)
    for (int p = 0; p < 3; p++) gemm_bf16_kernel<<<gg, 256, 0, stream>>>(ws, c, p); // edge GEMMs
    ew_kernel<<<RW, 256, 0, stream>>>(ws, c, 3);                                    // node pre / ew acts
    for (int s = 3; s < 6; s++) gemm_bf16_kernel<<<gg, 256, 0, stream>>>(ws, c, s); // projections
    ew_kernel<<<RW, 256, 0, stream>>>(ws, c, 4);                                    // GLU product
    vtrans_kernel<<<dim3(32, 16, 4), tb, 0, stream>>>(ws, c);                       // act0 only
    attn_kernel<<<256, 256, 0, stream>>>(ws, c);                                    // act0 only
    gemm_bf16_kernel<<<gg, 256, 0, stream>>>(ws, c, 6);                             // output proj
  }
  final_kernel<<<RW, 256, 0, stream>>>(ws, out_g, out_beta, (float*)d_out);

  (void)in_sizes; (void)n_in; (void)out_size; (void)ws_size;
}
